// MambaBlock_57801669870242
// MI455X (gfx1250) — compile-verified, matrix operand order checked by reading, not run on hardware
//
#include <hip/hip_runtime.h>

// ---------------------------------------------------------------------------
// Types for CDNA5 WMMA
// ---------------------------------------------------------------------------
typedef __attribute__((ext_vector_type(16))) __bf16  v16bf;
typedef __attribute__((ext_vector_type(8)))  float   v8f;
typedef __attribute__((ext_vector_type(4)))  unsigned v4u;

union FragU {
    v4u   q[2];
    v16bf v;
};

__device__ __forceinline__ unsigned short f32_to_bf16(float f) {
    unsigned u = __float_as_uint(f);
    unsigned rnd = 0x7FFFu + ((u >> 16) & 1u);
    return (unsigned short)((u + rnd) >> 16);
}

// Load a 16x32 bf16 WMMA fragment (A-layout; B-operand uses the same pattern
// on rows of W since W is (N,K) row-major and the GEMM computes A*W^T).
// Row-major source (rows, K), leading dim = K (compile-time, multiple of 32).
// Per-lane data = two contiguous 16B spans:
//   k in [kb + hi*8, +8)  and  [kb + 16 + hi*8, +8)     (hi = lane >= 16)
// -> two unconditional global_load_b128, no guards, no divergence.
template <int K>
__device__ __forceinline__ v16bf load_frag(const unsigned short* __restrict__ base,
                                           int row, int kb, int hi) {
    FragU f;
    const unsigned short* p = base + (long)row * K + kb + hi * 8;
    f.q[0] = *(const v4u*)(p);
    f.q[1] = *(const v4u*)(p + 16);
    return f.v;
}

// ---------------------------------------------------------------------------
// WMMA GEMM:  C[M,N] = A[M,K](bf16) * W[N,K](bf16)^T   (fp32 accum)
// EPI = 0: store acc
// EPI = 1: store softplus(acc + aux[n])          (dt projection)
// EPI = 2: store acc + aux[m*N + n]              (residual add, out-proj)
// One wave -> 32(M) x 64(N) tile: 2 M-tiles x 4 N-tiles, 8 wmma / K-step.
// All 6 fragment loads (2 A + 4 B = 12x global_load_b128) are issued as one
// batch before the 8 wmma, so the compiler emits a single load clause + one
// s_wait_loadcnt per K-step and can overlap next-iteration loads with wmma.
// ---------------------------------------------------------------------------
template <int K, int EPI>
__global__ __launch_bounds__(256)
void gemm_wmma(const unsigned short* __restrict__ A,
               const unsigned short* __restrict__ W,
               float* __restrict__ C,
               int M, int N,
               const float* __restrict__ aux) {
    const int lane = threadIdx.x & 31;
    const int wave = threadIdx.x >> 5;
    const int hi   = lane >> 4;
    const int lo   = lane & 15;

    const int Mtiles = M >> 5;              // 32-row tiles
    const int NG     = (N + 63) >> 6;
    long job = (long)blockIdx.x * (blockDim.x >> 5) + wave;
    if (job >= (long)Mtiles * NG) return;

    const int mt = (int)(job % Mtiles);
    const int ng = (int)(job / Mtiles);

    const int arow0 = mt * 32;
    const int rA0   = arow0 + lo;
    const int rA1   = arow0 + 16 + lo;

    int nrow[4];
#pragma unroll
    for (int j = 0; j < 4; ++j)
        nrow[j] = min((ng * 4 + j) * 16 + lo, N - 1);

    v8f acc[2][4] = {};

#pragma unroll 2
    for (int kb = 0; kb < K; kb += 32) {
        // Batch all operand loads: one clause of 12x global_load_b128.
        v16bf a0 = load_frag<K>(A, rA0, kb, hi);
        v16bf a1 = load_frag<K>(A, rA1, kb, hi);
        v16bf b0 = load_frag<K>(W, nrow[0], kb, hi);
        v16bf b1 = load_frag<K>(W, nrow[1], kb, hi);
        v16bf b2 = load_frag<K>(W, nrow[2], kb, hi);
        v16bf b3 = load_frag<K>(W, nrow[3], kb, hi);
        // Speculative prefetch of next K-step of A (global_prefetch_b8).
        __builtin_prefetch(A + (long)rA0 * K + kb + 32, 0, 1);

        acc[0][0] = __builtin_amdgcn_wmma_f32_16x16x32_bf16(
            false, a0, false, b0, (short)0, acc[0][0], false, false);
        acc[1][0] = __builtin_amdgcn_wmma_f32_16x16x32_bf16(
            false, a1, false, b0, (short)0, acc[1][0], false, false);
        acc[0][1] = __builtin_amdgcn_wmma_f32_16x16x32_bf16(
            false, a0, false, b1, (short)0, acc[0][1], false, false);
        acc[1][1] = __builtin_amdgcn_wmma_f32_16x16x32_bf16(
            false, a1, false, b1, (short)0, acc[1][1], false, false);
        acc[0][2] = __builtin_amdgcn_wmma_f32_16x16x32_bf16(
            false, a0, false, b2, (short)0, acc[0][2], false, false);
        acc[1][2] = __builtin_amdgcn_wmma_f32_16x16x32_bf16(
            false, a1, false, b2, (short)0, acc[1][2], false, false);
        acc[0][3] = __builtin_amdgcn_wmma_f32_16x16x32_bf16(
            false, a0, false, b3, (short)0, acc[0][3], false, false);
        acc[1][3] = __builtin_amdgcn_wmma_f32_16x16x32_bf16(
            false, a1, false, b3, (short)0, acc[1][3], false, false);
    }

#pragma unroll
    for (int mi = 0; mi < 2; ++mi) {
#pragma unroll
        for (int j = 0; j < 4; ++j) {
            int n = (ng * 4 + j) * 16 + lo;
            if (n >= N) continue;
#pragma unroll
            for (int r = 0; r < 8; ++r) {
                int m = arow0 + mi * 16 + r + hi * 8;
                float v = acc[mi][j][r];
                if (EPI == 1) {
                    v += aux[n];
                    v = (v > 20.0f) ? v : __logf(1.0f + __expf(v));
                } else if (EPI == 2) {
                    v += aux[(long)m * N + n];
                }
                C[(long)m * N + n] = v;
            }
        }
    }
}

// ---------------------------------------------------------------------------
// fp32 -> bf16 converters
// ---------------------------------------------------------------------------
__global__ void cvt_f32_bf16(const float* __restrict__ in,
                             unsigned short* __restrict__ out, long n) {
    long i = (long)blockIdx.x * blockDim.x + threadIdx.x;
    if (i < n) out[i] = f32_to_bf16(in[i]);
}

// Row-wise convert with K padding: in (rows, Kin) f32 -> out (rows, Kout) bf16,
// cols >= Kin zero-filled. Used for W_dt (48 -> 64).
__global__ void cvt_pad_bf16(const float* __restrict__ in,
                             unsigned short* __restrict__ out,
                             long rows, int Kin, int Kout) {
    long i = (long)blockIdx.x * blockDim.x + threadIdx.x;
    if (i >= rows * Kout) return;
    long r = i / Kout;
    int  c = (int)(i % Kout);
    out[i] = (c < Kin) ? f32_to_bf16(in[r * Kin + c]) : (unsigned short)0;
}

// dt slice of x_dbl (T x 80, cols [0,48)) -> bf16 (T x 64), cols 48..63 zero.
__global__ void cvt_dt_slice(const float* __restrict__ xdbl,
                             unsigned short* __restrict__ out, long rows) {
    long i = (long)blockIdx.x * blockDim.x + threadIdx.x;
    if (i >= rows * 64) return;
    long m = i / 64;
    int  c = (int)(i % 64);
    out[i] = (c < 48) ? f32_to_bf16(xdbl[m * 80 + c]) : (unsigned short)0;
}

// ---------------------------------------------------------------------------
// LayerNorm over D=768 per token -> bf16
// ---------------------------------------------------------------------------
__global__ __launch_bounds__(256)
void layernorm_bf16(const float* __restrict__ x,
                    const float* __restrict__ g,
                    const float* __restrict__ b,
                    unsigned short* __restrict__ h, int D) {
    __shared__ float red0[8];
    __shared__ float red1[8];
    const long base = (long)blockIdx.x * D;
    float s = 0.f, s2 = 0.f;
    for (int i = threadIdx.x; i < D; i += 256) {
        float v = x[base + i];
        s += v; s2 += v * v;
    }
#pragma unroll
    for (int off = 16; off > 0; off >>= 1) {
        s  += __shfl_down(s,  off);
        s2 += __shfl_down(s2, off);
    }
    if ((threadIdx.x & 31) == 0) {
        red0[threadIdx.x >> 5] = s;
        red1[threadIdx.x >> 5] = s2;
    }
    __syncthreads();
    if (threadIdx.x == 0) {
        float a = 0.f, c = 0.f;
#pragma unroll
        for (int i = 0; i < 8; ++i) { a += red0[i]; c += red1[i]; }
        float mu  = a / (float)D;
        float var = c / (float)D - mu * mu;
        red0[0] = mu;
        red1[0] = rsqrtf(var + 1e-5f);
    }
    __syncthreads();
    float mu = red0[0], rs = red1[0];
    for (int i = threadIdx.x; i < D; i += 256) {
        float v = (x[base + i] - mu) * rs * g[i] + b[i];
        h[base + i] = f32_to_bf16(v);
    }
}

// ---------------------------------------------------------------------------
// Depthwise causal conv1d (taps=4) + bias + SiLU -> fp32 + bf16
// xz layout: (B*L, 2*Di) fp32; xc channels are cols [0, Di)
// ---------------------------------------------------------------------------
__global__ __launch_bounds__(256)
void conv_silu(const float* __restrict__ xz,
               const float* __restrict__ Wc,
               const float* __restrict__ bc,
               float* __restrict__ xcf,
               unsigned short* __restrict__ xcb,
               int L, int Di) {
    long idx = (long)blockIdx.x * blockDim.x + threadIdx.x;
    int  d  = (int)(idx % Di);
    long bl = idx / Di;        // b*L + l
    int  l  = (int)(bl % L);
    float s = bc[d];
#pragma unroll
    for (int j = 0; j < 4; ++j) {
        int ls = l - 3 + j;
        if (ls >= 0)
            s += Wc[d * 4 + j] * xz[(bl - 3 + j) * (2L * Di) + d];
    }
    float sig = 1.0f / (1.0f + __expf(-s));
    float v = s * sig;
    xcf[idx] = v;
    xcb[idx] = f32_to_bf16(v);
}

// ---------------------------------------------------------------------------
// Selective scan: one lane per (b,d) channel, 16 states in registers.
// B/C (shared across all channels of a batch) staged in LDS, 64 steps/chunk.
// Fused epilogue: y = (y_scan + xc*D) * silu(z)  -> bf16 for out-proj.
// Grid = B * (Di/256) blocks of 256 threads.
// ---------------------------------------------------------------------------
__global__ __launch_bounds__(256)
void selective_scan(const float* __restrict__ dt,     // (B*L, Di)
                    const float* __restrict__ xdbl,   // (B*L, 80): B @48, C @64
                    const float* __restrict__ xcf,    // (B*L, Di)
                    const float* __restrict__ xz,     // (B*L, 2*Di): z @Di
                    const float* __restrict__ A_log,  // (Di, 16)
                    const float* __restrict__ Dp,     // (Di)
                    unsigned short* __restrict__ ybf, // (B*L, Di)
                    int L, int Di) {
    __shared__ float sB[64][16];
    __shared__ float sC[64][16];

    const int tid    = threadIdx.x;
    const int chunks = Di >> 8;                // blocks per batch
    const int b      = blockIdx.x / chunks;
    const int d      = (blockIdx.x % chunks) * 256 + tid;

    float h[16], Av[16];
#pragma unroll
    for (int n = 0; n < 16; ++n) {
        h[n]  = 0.0f;
        Av[n] = -__expf(A_log[(long)d * 16 + n]);
    }
    const float Dv = Dp[d];

    for (int l0 = 0; l0 < L; l0 += 64) {
        __syncthreads();
        for (int i = tid; i < 64 * 32; i += 256) {
            int t = i >> 5, c = i & 31;
            float v = xdbl[((long)b * L + l0 + t) * 80 + 48 + c];
            if (c < 16) sB[t][c]      = v;
            else        sC[t][c - 16] = v;
        }
        __syncthreads();

        for (int t = 0; t < 64; ++t) {
            long row = (long)b * L + l0 + t;
            long idx = row * Di + d;
            float dtv = dt[idx];
            float xv  = xcf[idx];
            float dtx = dtv * xv;
            float y   = 0.0f;
#pragma unroll
            for (int n = 0; n < 16; ++n) {
                float dA = __expf(dtv * Av[n]);
                h[n] = dA * h[n] + dtx * sB[t][n];
                y   += h[n] * sC[t][n];
            }
            float zv  = xz[row * (2L * Di) + Di + d];
            float sig = 1.0f / (1.0f + __expf(-zv));
            float out = (y + xv * Dv) * (zv * sig);
            ybf[idx] = f32_to_bf16(out);
        }
    }
}

// ---------------------------------------------------------------------------
// Host launcher
// ---------------------------------------------------------------------------
extern "C" void kernel_launch(void* const* d_in, const int* in_sizes, int n_in,
                              void* d_out, int out_size, void* d_ws, size_t ws_size,
                              hipStream_t stream) {
    (void)in_sizes; (void)n_in; (void)out_size; (void)ws_size;

    constexpr int B = 2, L = 4096, DM = 768, DI = 1536, N = 16, DTR = 48;
    constexpr int DTP = 64;                    // padded dt_rank for WMMA K
    constexpr long T = (long)B * L;            // 8192 tokens

    const float* x      = (const float*)d_in[0];
    const float* ln_g   = (const float*)d_in[1];
    const float* ln_b   = (const float*)d_in[2];
    const float* W_in   = (const float*)d_in[3];   // (3072, 768)
    const float* W_conv = (const float*)d_in[4];   // (1536, 4)
    const float* b_conv = (const float*)d_in[5];
    const float* W_xp   = (const float*)d_in[6];   // (80, 1536)
    const float* W_dt   = (const float*)d_in[7];   // (1536, 48)
    const float* b_dt   = (const float*)d_in[8];
    const float* A_log  = (const float*)d_in[9];   // (1536, 16)
    const float* D_par  = (const float*)d_in[10];
    const float* W_out  = (const float*)d_in[11];  // (768, 1536)
    float* out = (float*)d_out;

    // Workspace carve-up (256B aligned)
    char* ws = (char*)d_ws;
    size_t off = 0;
    auto carve = [&](size_t bytes) -> void* {
        void* p = ws + off;
        off += (bytes + 255) & ~(size_t)255;
        return p;
    };
    unsigned short* Win_bf  = (unsigned short*)carve((size_t)2 * DI * DM * 2);
    unsigned short* Wxp_bf  = (unsigned short*)carve((size_t)(DTR + 2 * N) * DI * 2);
    unsigned short* Wdt_bf  = (unsigned short*)carve((size_t)DI * DTP * 2);
    unsigned short* Wout_bf = (unsigned short*)carve((size_t)DM * DI * 2);
    unsigned short* h_bf    = (unsigned short*)carve((size_t)T * DM * 2);
    float*          xzbuf   = (float*)carve((size_t)T * 2 * DI * 4);
    float*          xcf     = (float*)carve((size_t)T * DI * 4);
    unsigned short* xcb     = (unsigned short*)carve((size_t)T * DI * 2);
    float*          xdbl    = (float*)carve((size_t)T * 80 * 4);
    unsigned short* dt_bf   = (unsigned short*)carve((size_t)T * DTP * 2);
    float*          dt_f    = (float*)carve((size_t)T * DI * 4);
    unsigned short* y_bf    = (unsigned short*)carve((size_t)T * DI * 2);

    auto cdiv = [](long a, long b) { return (unsigned)((a + b - 1) / b); };

    // 1) weights -> bf16 (W_dt zero-padded from K=48 to K=64)
    cvt_f32_bf16<<<cdiv(2L * DI * DM, 256), 256, 0, stream>>>(W_in, Win_bf, 2L * DI * DM);
    cvt_f32_bf16<<<cdiv((long)(DTR + 2 * N) * DI, 256), 256, 0, stream>>>(W_xp, Wxp_bf, (long)(DTR + 2 * N) * DI);
    cvt_pad_bf16<<<cdiv((long)DI * DTP, 256), 256, 0, stream>>>(W_dt, Wdt_bf, (long)DI, DTR, DTP);
    cvt_f32_bf16<<<cdiv((long)DM * DI, 256), 256, 0, stream>>>(W_out, Wout_bf, (long)DM * DI);

    // 2) LayerNorm -> bf16
    layernorm_bf16<<<(unsigned)T, 256, 0, stream>>>(x, ln_g, ln_b, h_bf, DM);

    // 3) xz = h @ W_in^T   (M=8192, N=3072, K=768)
    {
        long jobs = (T / 32) * ((2 * DI + 63) / 64);
        gemm_wmma<DM, 0><<<cdiv(jobs, 8), 256, 0, stream>>>(
            h_bf, Win_bf, xzbuf, (int)T, 2 * DI, nullptr);
    }

    // 4) depthwise causal conv + SiLU
    conv_silu<<<cdiv(T * DI, 256), 256, 0, stream>>>(xzbuf, W_conv, b_conv, xcf, xcb, L, DI);

    // 5) x_dbl = xc @ W_xproj^T   (M=8192, N=80, K=1536)
    {
        long jobs = (T / 32) * ((80 + 63) / 64);
        gemm_wmma<DI, 0><<<cdiv(jobs, 8), 256, 0, stream>>>(
            xcb, Wxp_bf, xdbl, (int)T, DTR + 2 * N, nullptr);
    }

    // 6) dt slice -> bf16, padded to 64 cols
    cvt_dt_slice<<<cdiv(T * DTP, 256), 256, 0, stream>>>(xdbl, dt_bf, T);

    // 7) dt = softplus(dt_in @ W_dt^T + b_dt)   (M=8192, N=1536, K=64)
    {
        long jobs = (T / 32) * ((DI + 63) / 64);
        gemm_wmma<DTP, 1><<<cdiv(jobs, 8), 256, 0, stream>>>(
            dt_bf, Wdt_bf, dt_f, (int)T, DI, b_dt);
    }

    // 8) selective scan + fused gate/skip epilogue -> y (bf16)
    selective_scan<<<B * (DI / 256), 256, 0, stream>>>(
        dt_f, xdbl, xcf, xzbuf, A_log, D_par, y_bf, L, DI);

    // 9) out = y @ W_out^T + x   (M=8192, N=768, K=1536), residual fused
    {
        long jobs = (T / 32) * ((DM + 63) / 64);
        gemm_wmma<DI, 2><<<cdiv(jobs, 8), 256, 0, stream>>>(
            y_bf, Wout_bf, out, (int)T, DM, x);
    }
}
